// KBRDModel_62715112456323
// MI455X (gfx1250) — matrix operand order repaired and checked by reading, not compiled, not run on hardware
//
#include <hip/hip_runtime.h>
#include <hip/hip_bf16.h>
#include <math.h>

#define B_ 2048
#define L_ 128
#define V_ 50000
#define D_ 128
#define HSTR 136   // padded LDS row stride in bf16 elements (272B = 68 dwords: bank-skewed, 16B-aligned)

typedef __attribute__((ext_vector_type(16))) __bf16        v16bf;
typedef __attribute__((ext_vector_type(8)))  float         v8f;
typedef __attribute__((ext_vector_type(8)))  unsigned int  v8u;
typedef __attribute__((ext_vector_type(4)))  unsigned int  v4u;
typedef __attribute__((ext_vector_type(4)))  float         v4f;
typedef __attribute__((ext_vector_type(4)))  unsigned int  u32x4;
typedef __attribute__((ext_vector_type(8)))  int           i32x8;
typedef __attribute__((ext_vector_type(4)))  int           i32x4;

#if __has_builtin(__builtin_amdgcn_tensor_load_to_lds) && __has_builtin(__builtin_amdgcn_s_wait_tensorcnt)
#define USE_TDM 1
#else
#define USE_TDM 0
#endif

__device__ __forceinline__ unsigned short f2bf(float f) {
  unsigned int u = __float_as_uint(f);
  u += 0x7FFFu + ((u >> 16) & 1u);           // round-to-nearest-even
  return (unsigned short)(u >> 16);
}
__device__ __forceinline__ float bf2f(unsigned short h) {
  return __uint_as_float(((unsigned int)h) << 16);
}
__device__ __forceinline__ float fast_tanh(float x) {
#if __has_builtin(__builtin_amdgcn_tanhf)
  return __builtin_amdgcn_tanhf(x);          // V_TANH_F32 (CDNA5 trans op)
#else
  return tanhf(x);
#endif
}

// ---------- kernel 0a: emb fp32 -> bf16 (one-time; 12.8MB, L2-resident afterwards) ----------
__global__ __launch_bounds__(256)
void k_convert_emb(const float* __restrict__ src, unsigned short* __restrict__ dst, int n) {
  int i = (blockIdx.x * 256 + threadIdx.x) * 4;
  if (i + 3 < n) {
    v4f f = *(const v4f*)(src + i);
    unsigned int p0 = (unsigned int)f2bf(f.x) | ((unsigned int)f2bf(f.y) << 16);
    unsigned int p1 = (unsigned int)f2bf(f.z) | ((unsigned int)f2bf(f.w) << 16);
    *(unsigned int*)(dst + i)     = p0;
    *(unsigned int*)(dst + i + 2) = p1;
  }
}

// ---------- kernel 0b: At[n][k] = attn_a[k][n] in bf16 (contiguous WMMA B-operand loads) ----------
__global__ __launch_bounds__(256)
void k_transpose_a(const float* __restrict__ A, unsigned short* __restrict__ At) {
  for (int i = threadIdx.x; i < D_ * D_; i += 256) {
    int n = i / D_, k = i % D_;
    At[n * D_ + k] = f2bf(A[k * D_ + n]);
  }
}

// ---------- kernel 1: per-batch-row attention -> user vector (bf16) ----------
__global__ __launch_bounds__(256)
void k_user(const int* __restrict__ ids, const int* __restrict__ mask,
            const float* __restrict__ emb, const unsigned short* __restrict__ At,
            const float* __restrict__ bvec, unsigned short* __restrict__ userb) {
  __shared__ __attribute__((aligned(16))) unsigned short h[L_ * HSTR];
  __shared__ float e_lds[L_];
  __shared__ float attn_lds[L_];

  const int b = blockIdx.x;
  const int t = threadIdx.x;

  if (t < L_) e_lds[t] = 0.f;

  // gather h rows, converting fp32 -> bf16 into LDS (2 threads per row)
  {
    int l = t >> 1, half = t & 1;
    int id = ids[b * L_ + l];
    if (id < 0) id = 0;
    if (id >= V_) id = V_ - 1;
    const float* src = emb + (long)id * D_ + half * 64;
    unsigned int* dst = (unsigned int*)(h + l * HSTR + half * 64);
#pragma unroll
    for (int i = 0; i < 16; ++i) {
      v4f f = *(const v4f*)(src + i * 4);
      dst[i * 2 + 0] = (unsigned int)f2bf(f.x) | ((unsigned int)f2bf(f.y) << 16);
      dst[i * 2 + 1] = (unsigned int)f2bf(f.z) | ((unsigned int)f2bf(f.w) << 16);
    }
  }
  __syncthreads();

  // T = tanh(h @ A); e = T @ bvec  (WMMA bf16, 8 waves x 16 M-rows each)
  const int wave = t >> 5, lane = t & 31;
  const int m0 = wave * 16;
  const int lrow = lane & 15, lhi = lane >> 4;

  for (int nt = 0; nt < 8; ++nt) {
    const int n0 = nt * 16;
    const int col = n0 + lrow;
    v8f acc = {};
#pragma unroll
    for (int kc = 0; kc < 4; ++kc) {
      const int kk = kc * 32;
      const unsigned short* hp = h + (m0 + lrow) * HSTR + kk + lhi * 16;
      v4u a_lo = *(const v4u*)(hp);
      v4u a_hi = *(const v4u*)(hp + 8);
      v8u a8;
      a8[0] = a_lo[0]; a8[1] = a_lo[1]; a8[2] = a_lo[2]; a8[3] = a_lo[3];
      a8[4] = a_hi[0]; a8[5] = a_hi[1]; a8[6] = a_hi[2]; a8[7] = a_hi[3];
      v16bf amat = __builtin_bit_cast(v16bf, a8);
      v8u b8 = *(const v8u*)(At + col * D_ + kk + lhi * 16);
      v16bf bmat = __builtin_bit_cast(v16bf, b8);
      acc = __builtin_amdgcn_wmma_f32_16x16x32_bf16(false, amat, false, bmat,
                                                    (short)0, acc, false, false);
    }
    const float bv = bvec[col];
#pragma unroll
    for (int r = 0; r < 8; ++r) {
      float tv = fast_tanh(acc[r]);
      atomicAdd(&e_lds[m0 + lhi * 8 + r], tv * bv);   // ds_add_f32 reduction over N
    }
  }
  __syncthreads();

  if (t < L_) {
    float e = e_lds[t];
    float a = 1.f / (1.f + __expf(-e));
    attn_lds[t] = a * (float)mask[b * L_ + t];
  }
  __syncthreads();

  // user[d] = sum_l attn[l] * h[l][d]
  if (t < D_) {
    float sum = 0.f;
#pragma unroll 4
    for (int l = 0; l < L_; ++l)
      sum += attn_lds[l] * bf2f(h[l * HSTR + t]);
    userb[b * D_ + t] = f2bf(sum);
  }
}

// ---------- kernel 2: logits = user @ emb^T + bias  (M=2048, N=50000, K=128) ----------
// WG tile: 128M x 128N. B tile (128 cols x 128 K, bf16) staged in LDS once per WG via the
// Tensor Data Mover (padded to 272B row stride via D# pad fields); 8 waves each own 16 M rows.
__global__ __launch_bounds__(256)
void k_logits(const unsigned short* __restrict__ userb,
              const unsigned short* __restrict__ embb,
              const float* __restrict__ bias,
              float* __restrict__ out) {
  __shared__ __attribute__((aligned(16))) unsigned short hB[128 * HSTR];

  const int t = threadIdx.x;
  const int wave = t >> 5, lane = t & 31;
  const int lrow = lane & 15, lhi = lane >> 4;
  const int nbase = blockIdx.x * 128;
  const int m0 = blockIdx.y * 128 + wave * 16;

#if USE_TDM
  if (wave == 0) {
    // D# per CDNA5 ISA. data_size=8B units.
    // Tile = 32 (x, 8B units = 256B row) x 128 rows; tensor_dim1 = V - nbase so the
    // TDM zero-fills out-of-range rows at the N edge. Pad: 4 dwords after every 64
    // dwords -> 272B LDS row stride (matches HSTR).
    unsigned lds_base = (unsigned)(unsigned long long)(&hB[0]);
    unsigned long long ga = (unsigned long long)(embb + (unsigned long long)nbase * D_);
    int rows = V_ - nbase; if (rows > 128) rows = 128;
    u32x4 g0;
    g0[0] = 1u;                                              // count=1, user mode
    g0[1] = lds_base;                                        // lds_addr
    g0[2] = (unsigned)ga;                                    // global_addr[31:0]
    g0[3] = (unsigned)((ga >> 32) & 0x01FFFFFFu) | (2u << 30); // global_addr[56:32], type=2
    i32x8 g1;
    g1[0] = (int)((3u << 16) | (1u << 20) | (5u << 22) | (3u << 25)); // data_size=8B, pad_en, interval=64dw, amount=4dw
    g1[1] = (int)(32u << 16);                                // tensor_dim0[15:0]=32 (8B units)
    g1[2] = (int)(((unsigned)rows & 0xFFFFu) << 16);         // tensor_dim0 hi=0; tensor_dim1[15:0]=rows
    g1[3] = (int)(32u << 16);                                // tensor_dim1 hi=0; tile_dim0=32
    g1[4] = (int)128u;                                       // tile_dim1=128; tile_dim2=0
    g1[5] = (int)32u;                                        // tensor_dim0_stride[31:0]=32 (8B units)
    g1[6] = 0;                                               // stride0 hi, stride1 lo
    g1[7] = 0;
    i32x4 z4 = {0, 0, 0, 0};
    i32x8 z8 = {0, 0, 0, 0, 0, 0, 0, 0};
    // 6-arg form (this toolchain): (g0, g1, grp2, grp3, grp3b, cpol)
    __builtin_amdgcn_tensor_load_to_lds(g0, g1, z4, z4, z8, 0);
    __builtin_amdgcn_s_wait_tensorcnt(0);
  }
#else
  // manual staging fallback: 2048 x 16B chunks, 8 per thread
  for (int i = t; i < 128 * 16; i += 256) {
    int row = i >> 4, c = i & 15;
    int v = nbase + row;
    int vc = v < V_ ? v : 0;
    v4u d = *(const v4u*)(embb + (unsigned long long)vc * D_ + c * 8);
    *(v4u*)(hB + row * HSTR + c * 8) = d;
  }
#endif
  __syncthreads();

  v8f acc[8] = {};
#pragma unroll
  for (int kc = 0; kc < 4; ++kc) {
    const int kk = kc * 32;
    // A-operand (user rows, bf16) from global: reused across all 8 N-tiles
    const unsigned short* up = userb + (m0 + lrow) * D_ + kk + lhi * 16;
    v4u a_lo = *(const v4u*)(up);
    v4u a_hi = *(const v4u*)(up + 8);
    v8u a8;
    a8[0] = a_lo[0]; a8[1] = a_lo[1]; a8[2] = a_lo[2]; a8[3] = a_lo[3];
    a8[4] = a_hi[0]; a8[5] = a_hi[1]; a8[6] = a_hi[2]; a8[7] = a_hi[3];
    v16bf amat = __builtin_bit_cast(v16bf, a8);
#pragma unroll
    for (int nt = 0; nt < 8; ++nt) {
      const unsigned short* bp = hB + (nt * 16 + lrow) * HSTR + kk + lhi * 16;
      v8u b8 = *(const v8u*)(bp);                      // 2x ds_load_b128, bank-optimal
      v16bf bmat = __builtin_bit_cast(v16bf, b8);
      acc[nt] = __builtin_amdgcn_wmma_f32_16x16x32_bf16(false, amat, false, bmat,
                                                        (short)0, acc[nt], false, false);
    }
  }
#pragma unroll
  for (int nt = 0; nt < 8; ++nt) {
    const int col = nbase + nt * 16 + lrow;
    if (col < V_) {
      const float bv = bias[col];
      const int row = m0 + lhi * 8;
#pragma unroll
      for (int r = 0; r < 8; ++r)
        out[(long)(row + r) * V_ + col] = acc[nt][r] + bv;
    }
  }
}

extern "C" void kernel_launch(void* const* d_in, const int* in_sizes, int n_in,
                              void* d_out, int out_size, void* d_ws, size_t ws_size,
                              hipStream_t stream) {
  (void)in_sizes; (void)n_in; (void)out_size; (void)ws_size;
  const int*   ids      = (const int*)d_in[0];
  const int*   mask     = (const int*)d_in[1];
  const float* emb      = (const float*)d_in[2];
  const float* attn_a   = (const float*)d_in[3];
  const float* attn_b   = (const float*)d_in[4];
  const float* rec_bias = (const float*)d_in[5];
  float* out = (float*)d_out;

  char* ws = (char*)d_ws;
  unsigned short* embb  = (unsigned short*)ws;                                        // V*D bf16
  unsigned short* Atb   = (unsigned short*)(ws + (size_t)V_ * D_ * 2);                // D*D bf16
  unsigned short* userb = (unsigned short*)(ws + (size_t)V_ * D_ * 2 + D_ * D_ * 2);  // B*D bf16

  k_convert_emb<<<dim3((V_ * D_ / 4 + 255) / 256), dim3(256), 0, stream>>>(emb, embb, V_ * D_);
  k_transpose_a<<<dim3(1), dim3(256), 0, stream>>>(attn_a, Atb);
  k_user<<<dim3(B_), dim3(256), 0, stream>>>(ids, mask, emb, Atb, attn_b, userb);
  k_logits<<<dim3((V_ + 127) / 128, B_ / 128), dim3(256), 0, stream>>>(userb, embb, rec_bias, out);
}
